// EGNN_dynamics_LP_27436251087301
// MI455X (gfx1250) — matrix operand order repaired and checked by reading, not compiled, not run on hardware
//
#include <hip/hip_runtime.h>

typedef __attribute__((ext_vector_type(16))) _Float16 v16h;
typedef __attribute__((ext_vector_type(8)))  _Float16 v8h;
typedef __attribute__((ext_vector_type(8)))  float    v8f;
typedef __attribute__((ext_vector_type(4)))  int      v4i;

#define BSZ  64
#define NN   64
#define BN   4096      // BSZ*NN
#define NE   262144    // BN*NN
#define HID  128
#define HPAD 136       // padded f16 row (272B, 16B aligned, bank-conflict free)

#if __has_builtin(__builtin_amdgcn_global_load_async_to_lds_b128) && \
    __has_builtin(__builtin_amdgcn_s_wait_asynccnt)
#define HAVE_ASYNC_LDS 1
#endif

#ifdef HAVE_ASYNC_LDS
typedef __attribute__((address_space(1))) v4i gv4i_t;   // global int4
typedef __attribute__((address_space(3))) v4i lv4i_t;   // LDS int4
__device__ __forceinline__ void async_copy16(const void* g, void* l) {
  // global -> LDS 128-bit async copy, tracked by ASYNCcnt
  __builtin_amdgcn_global_load_async_to_lds_b128((gv4i_t*)g, (lv4i_t*)l, 0, 0);
}
#endif

__device__ __forceinline__ float silu_f(float x) { return x / (1.0f + __expf(-x)); }

__device__ __forceinline__ v16h load_frag16(const _Float16* lo, const _Float16* hi) {
  v8h a = *(const v8h*)lo;
  v8h b = *(const v8h*)hi;
  v16h r;
#pragma unroll
  for (int e = 0; e < 8; ++e) { r[e] = a[e]; r[e + 8] = b[e]; }
  return r;
}

// Stage pre-padded f16 W2T (HID*HPAD halfwords) from global into LDS.
__device__ __forceinline__ void stage_w2t(const _Float16* __restrict__ W2T,
                                          _Float16* __restrict__ lds, int tid) {
  const v8h* gs = (const v8h*)W2T;
  v8h* ld = (v8h*)lds;
#ifdef HAVE_ASYNC_LDS
  for (int c = tid; c < (HID * HPAD) / 8; c += 256) async_copy16(gs + c, ld + c);
#else
  for (int c = tid; c < (HID * HPAD) / 8; c += 256) ld[c] = gs[c];
#endif
}

__device__ __forceinline__ void wait_w2t() {
#ifdef HAVE_ASYNC_LDS
  __builtin_amdgcn_s_wait_asynccnt(0);
#endif
}

// ---------------- prep: W2T[n][k] = (f16) W2[k][n], padded rows ----------------
__global__ void k_prep_w2t(const float* __restrict__ W2, _Float16* __restrict__ out) {
  int idx = blockIdx.x * blockDim.x + threadIdx.x;   // HID*HPAD
  if (idx >= HID * HPAD) return;
  int n = idx / HPAD, k = idx - n * HPAD;
  out[idx] = (k < HID) ? (_Float16)W2[k * HID + n] : (_Float16)0.0f;
}

// ---------------- setup: x0 = x = xh[:, :3] * nm ----------------
__global__ void k_setup_x(const float* __restrict__ xh, const float* __restrict__ nm,
                          float* __restrict__ x0, float* __restrict__ x) {
  int idx = blockIdx.x * blockDim.x + threadIdx.x;
  if (idx >= BN * 3) return;
  int n = idx / 3, d = idx % 3;
  float v = xh[n * 11 + d] * nm[n];
  x0[idx] = v;
  x[idx]  = v;
}

// ---------------- embed: h = [xh[:,3:]*nm, t, ctx] @ We + be ----------------
__global__ void k_embed(const float* __restrict__ xh, const float* __restrict__ nm,
                        const float* __restrict__ t, const float* __restrict__ ctx,
                        const float* __restrict__ W, const float* __restrict__ b,
                        float* __restrict__ h) {
  int idx = blockIdx.x * blockDim.x + threadIdx.x;   // BN*HID
  int n = idx >> 7, k = idx & 127;
  float nmv = nm[n];
  float s = b[k];
#pragma unroll
  for (int c = 0; c < 8; ++c) s += xh[n * 11 + 3 + c] * nmv * W[c * HID + k];
  s += t[0] * W[8 * HID + k];
  for (int c = 0; c < 32; ++c) s += ctx[n * 32 + c] * W[(9 + c) * HID + k];
  h[idx] = s;
}

// ---------------- dist + normalized coord_diff per edge ----------------
__global__ void k_dist(const float* __restrict__ x, const int* __restrict__ rows,
                       const int* __restrict__ cols, float* __restrict__ dist,
                       float* __restrict__ cd) {
  int e = blockIdx.x * blockDim.x + threadIdx.x;
  if (e >= NE) return;
  int r = rows[e], c = cols[e];
  float d0 = x[r * 3 + 0] - x[c * 3 + 0];
  float d1 = x[r * 3 + 1] - x[c * 3 + 1];
  float d2 = x[r * 3 + 2] - x[c * 3 + 2];
  float radial = d0 * d0 + d1 * d1 + d2 * d2;
  dist[e] = radial;
  float inv = 1.0f / sqrtf(radial + 1e-8f);
  cd[e * 3 + 0] = d0 * inv;
  cd[e * 3 + 1] = d1 * inv;
  cd[e * 3 + 2] = d2 * inv;
}

// ---------------- node-level 128x128 projection: out = h @ W (+ bias) ----------------
__global__ void k_lin128(const float* __restrict__ h, const float* __restrict__ W,
                         const float* __restrict__ bias, float* __restrict__ out) {
  int idx = blockIdx.x * blockDim.x + threadIdx.x;   // BN*HID
  int n = idx >> 7, k = idx & 127;
  float s = bias ? bias[k] : 0.0f;
  const float* hr = h + n * HID;
  for (int c = 0; c < HID; ++c) s += hr[c] * W[c * HID + k];
  out[idx] = s;
}

// ---------------- fused edge MLP layer2 + mask + segment reduce (WMMA) ----------------
// One WG per (b,i): hidden[j,k] = silu(P[i,k]+Q[b*64+j,k]+dist*wd[k]+dist0*wd0[k])
// m = silu(hidden @ W2 + b2) * em ; agg[i,:] = sum_j m / 100
__global__ __launch_bounds__(256) void k_edge_msg(
    const float* __restrict__ P, const float* __restrict__ Q,
    const float* __restrict__ dist, const float* __restrict__ dist0,
    const float* __restrict__ wd, const float* __restrict__ wd0,
    const _Float16* __restrict__ W2T, const float* __restrict__ b2,
    const float* __restrict__ em, float* __restrict__ agg) {
  __shared__ _Float16 w2t[HID][HPAD];   // transposed: w2t[n][k] = W2[k][n]
  __shared__ _Float16 hid[NN][HPAD];
  __shared__ float p_s[HID], b2_s[HID], wd_s[HID], wd0_s[HID], agg_s[HID];
  __shared__ float dist_s[NN], dist0_s[NN], em_s[NN];
  const int node  = blockIdx.x;          // b*64 + i
  const int qbase = node & ~(NN - 1);    // b*64
  const int tid   = threadIdx.x;

  stage_w2t(W2T, &w2t[0][0], tid);       // async DMA overlaps with hidden build

  if (tid < HID) {
    p_s[tid]   = P[node * HID + tid];
    b2_s[tid]  = b2[tid];
    wd_s[tid]  = wd[tid];
    wd0_s[tid] = wd0[tid];
    agg_s[tid] = 0.0f;
  }
  if (tid < NN) {
    int e = node * NN + tid;
    dist_s[tid]  = dist[e];
    dist0_s[tid] = dist0[e];
    em_s[tid]    = em[e];
  }
  __syncthreads();

  for (int idx = tid; idx < NN * HID; idx += 256) {
    int j = idx >> 7, k = idx & 127;
    float v = p_s[k] + Q[(qbase + j) * HID + k] + dist_s[j] * wd_s[k] + dist0_s[j] * wd0_s[k];
    hid[j][k] = (_Float16)silu_f(v);
  }
  wait_w2t();
  __syncthreads();

  const int wave = tid >> 5, lane = tid & 31;
  const int jt = wave >> 1;            // j tile 0..3 (16 rows each)
  const int ob = (wave & 1) * 4;       // base out tile (4 tiles of 16 cols)
  const int m = lane & 15, half = lane >> 4, bn = lane & 15;
  const int row = jt * 16 + m;

  v8f acc[4] = {};
#pragma unroll
  for (int kt = 0; kt < 4; ++kt) {
    const int kb = kt * 32;
    v16h a = load_frag16(&hid[row][kb + half * 8], &hid[row][kb + 16 + half * 8]);
#pragma unroll
    for (int ot = 0; ot < 4; ++ot) {
      const _Float16* bp = &w2t[(ob + ot) * 16 + bn][kb + half * 8];
      v16h bf = load_frag16(bp, bp + 16);
      acc[ot] = __builtin_amdgcn_wmma_f32_16x16x32_f16(false, a, false, bf,
                                                       (short)0, acc[ot], false, false);
    }
  }

#pragma unroll
  for (int ot = 0; ot < 4; ++ot) {
    const int ncol = (ob + ot) * 16 + bn;
    float partial = 0.0f;
#pragma unroll
    for (int g = 0; g < 8; ++g) {
      int j = jt * 16 + g + half * 8;
      partial += silu_f(acc[ot][g] + b2_s[ncol]) * em_s[j];
    }
    atomicAdd(&agg_s[ncol], partial);
  }
  __syncthreads();
  if (tid < HID) agg[node * HID + tid] = agg_s[tid] / 100.0f;
}

// ---------------- node MLP ----------------
__global__ void k_node_a(const float* __restrict__ h, const float* __restrict__ agg,
                         const float* __restrict__ W, const float* __restrict__ b,
                         float* __restrict__ tmp) {
  int idx = blockIdx.x * blockDim.x + threadIdx.x;   // BN*HID
  int n = idx >> 7, k = idx & 127;
  float s = b[k];
  for (int c = 0; c < HID; ++c) s += h[n * HID + c] * W[c * HID + k];
  for (int c = 0; c < HID; ++c) s += agg[n * HID + c] * W[(HID + c) * HID + k];
  tmp[idx] = silu_f(s);
}

__global__ void k_node_b(const float* __restrict__ h, const float* __restrict__ tmp,
                         const float* __restrict__ W, const float* __restrict__ b,
                         const float* __restrict__ nm, float* __restrict__ hout) {
  int idx = blockIdx.x * blockDim.x + threadIdx.x;
  int n = idx >> 7, k = idx & 127;
  float s = b[k];
  for (int c = 0; c < HID; ++c) s += tmp[n * HID + c] * W[c * HID + k];
  hout[idx] = (h[idx] + s) * nm[n];
}

// ---------------- fused coord MLP layers 2-3 + position update (WMMA) ----------------
__global__ __launch_bounds__(256) void k_coord(
    const float* __restrict__ P, const float* __restrict__ Q,
    const float* __restrict__ dist, const float* __restrict__ dist0,
    const float* __restrict__ wd, const float* __restrict__ wd0,
    const _Float16* __restrict__ W2T, const float* __restrict__ b2,
    const float* __restrict__ w3, const float* __restrict__ em,
    const float* __restrict__ cd, float* __restrict__ x) {
  __shared__ _Float16 w2t[HID][HPAD];
  __shared__ _Float16 hid[NN][HPAD];
  __shared__ float p_s[HID], b2_s[HID], wd_s[HID], wd0_s[HID], w3_s[HID];
  __shared__ float dist_s[NN], dist0_s[NN], em_s[NN], scalar_s[NN];
  const int node  = blockIdx.x;
  const int qbase = node & ~(NN - 1);
  const int tid   = threadIdx.x;

  stage_w2t(W2T, &w2t[0][0], tid);

  if (tid < HID) {
    p_s[tid]   = P[node * HID + tid];
    b2_s[tid]  = b2[tid];
    wd_s[tid]  = wd[tid];
    wd0_s[tid] = wd0[tid];
    w3_s[tid]  = w3[tid];
  }
  if (tid < NN) {
    int e = node * NN + tid;
    dist_s[tid]  = dist[e];
    dist0_s[tid] = dist0[e];
    em_s[tid]    = em[e];
  }
  __syncthreads();

  for (int idx = tid; idx < NN * HID; idx += 256) {
    int j = idx >> 7, k = idx & 127;
    float v = p_s[k] + Q[(qbase + j) * HID + k] + dist_s[j] * wd_s[k] + dist0_s[j] * wd0_s[k];
    hid[j][k] = (_Float16)silu_f(v);
  }
  wait_w2t();
  __syncthreads();

  const int wave = tid >> 5, lane = tid & 31;
  const int jt = wave >> 1;
  const int ob = (wave & 1) * 4;
  const int m = lane & 15, half = lane >> 4, bn = lane & 15;
  const int row = jt * 16 + m;

  v8f acc[4] = {};
#pragma unroll
  for (int kt = 0; kt < 4; ++kt) {
    const int kb = kt * 32;
    v16h a = load_frag16(&hid[row][kb + half * 8], &hid[row][kb + 16 + half * 8]);
#pragma unroll
    for (int ot = 0; ot < 4; ++ot) {
      const _Float16* bp = &w2t[(ob + ot) * 16 + bn][kb + half * 8];
      v16h bf = load_frag16(bp, bp + 16);
      acc[ot] = __builtin_amdgcn_wmma_f32_16x16x32_f16(false, a, false, bf,
                                                       (short)0, acc[ot], false, false);
    }
  }
  __syncthreads();   // all reads of hid complete; safe to overwrite with hidden2

#pragma unroll
  for (int ot = 0; ot < 4; ++ot) {
    const int ncol = (ob + ot) * 16 + bn;
#pragma unroll
    for (int g = 0; g < 8; ++g) {
      int j = jt * 16 + g + half * 8;
      hid[j][ncol] = (_Float16)silu_f(acc[ot][g] + b2_s[ncol]);
    }
  }
  __syncthreads();

  if (tid < NN) {
    float s = 0.0f;
    for (int k = 0; k < HID; ++k) s += (float)hid[tid][k] * w3_s[k];
    scalar_s[tid] = s * em_s[tid];
  }
  __syncthreads();

  if (tid < 3) {
    float a = 0.0f;
    for (int j = 0; j < NN; ++j) a += cd[(node * NN + j) * 3 + tid] * scalar_s[j];
    x[node * 3 + tid] += a / 100.0f;
  }
}

// ---------------- output head: first 8 cols of (h @ Wout + bout) * nm ----------------
__global__ void k_out_h(const float* __restrict__ h, const float* __restrict__ W,
                        const float* __restrict__ b, const float* __restrict__ nm,
                        float* __restrict__ h8) {
  int idx = blockIdx.x * blockDim.x + threadIdx.x;
  if (idx >= BN * 8) return;
  int n = idx >> 3, k = idx & 7;
  float s = b[k];
  for (int c = 0; c < HID; ++c) s += h[n * HID + c] * W[c * 41 + k];
  h8[idx] = s * nm[n];
}

// ---------------- velocity centering + concat ----------------
__global__ __launch_bounds__(64) void k_finalize(const float* __restrict__ x,
                                                 const float* __restrict__ x0,
                                                 const float* __restrict__ h8,
                                                 const float* __restrict__ nm,
                                                 float* __restrict__ out) {
  int b = blockIdx.x, n = threadIdx.x;
  int node = b * NN + n;
  __shared__ float vs[NN][3];
  __shared__ float nms[NN];
  __shared__ float mean[3];
  float nmv = nm[node];
  for (int d = 0; d < 3; ++d) vs[n][d] = (x[node * 3 + d] - x0[node * 3 + d]) * nmv;
  nms[n] = nmv;
  __syncthreads();
  if (n == 0) {
    float s0 = 0, s1 = 0, s2 = 0, sn = 0;
    for (int j = 0; j < NN; ++j) { s0 += vs[j][0]; s1 += vs[j][1]; s2 += vs[j][2]; sn += nms[j]; }
    mean[0] = s0 / sn; mean[1] = s1 / sn; mean[2] = s2 / sn;
  }
  __syncthreads();
  for (int d = 0; d < 3; ++d) out[node * 11 + d] = (vs[n][d] - mean[d]) * nmv;
  for (int k = 0; k < 8; ++k) out[node * 11 + 3 + k] = h8[node * 8 + k];
}

extern "C" void kernel_launch(void* const* d_in, const int* in_sizes, int n_in,
                              void* d_out, int out_size, void* d_ws, size_t ws_size,
                              hipStream_t stream) {
  const float* t   = (const float*)d_in[0];
  const float* xh  = (const float*)d_in[1];
  const float* nm  = (const float*)d_in[2];
  const float* em  = (const float*)d_in[3];
  const float* ctx = (const float*)d_in[4];
  const int*  rows = (const int*)d_in[5];
  const int*  cols = (const int*)d_in[6];

  const float* prm[88];
  for (int i = 0; i < 88 && (7 + i) < n_in; ++i) prm[i] = (const float*)d_in[7 + i];
  const float* embed_w = prm[0];
  const float* embed_b = prm[1];
  const float* out_w   = prm[2];
  const float* out_b   = prm[3];

  float* ws    = (float*)d_ws;
  float* x0v   = ws; ws += BN * 3;
  float* xv    = ws; ws += BN * 3;
  float* hA    = ws; ws += BN * HID;
  float* hB    = ws; ws += BN * HID;
  float* Pb    = ws; ws += BN * HID;
  float* Qb    = ws; ws += BN * HID;
  float* aggb  = ws; ws += BN * HID;
  float* tmpb  = ws; ws += BN * HID;
  float* dist0 = ws; ws += NE;
  float* dist  = ws; ws += NE;
  float* cd    = ws; ws += NE * 3;
  float* h8    = ws; ws += BN * 8;
  _Float16* w2tg = (_Float16*)ws; ws += (HID * HPAD) / 2;   // 16B-aligned staging buf

  const int PREP_GRID = (HID * HPAD + 255) / 256;

  k_setup_x<<<(BN * 3 + 255) / 256, 256, 0, stream>>>(xh, nm, x0v, xv);
  k_embed<<<BN * HID / 256, 256, 0, stream>>>(xh, nm, t, ctx, embed_w, embed_b, hA);
  k_dist<<<NE / 256, 256, 0, stream>>>(x0v, rows, cols, dist0, cd);

  float* hcur = hA;
  float* hnext = hB;
  for (int blk = 0; blk < 4; ++blk) {
    const float* const* bp = prm + 4 + blk * 21;
    k_dist<<<NE / 256, 256, 0, stream>>>(xv, rows, cols, dist, cd);
    for (int g = 0; g < 2; ++g) {
      const float* e0w = bp[g * 8 + 0];
      const float* e0b = bp[g * 8 + 1];
      const float* e1w = bp[g * 8 + 2];
      const float* e1b = bp[g * 8 + 3];
      const float* n0w = bp[g * 8 + 4];
      const float* n0b = bp[g * 8 + 5];
      const float* n1w = bp[g * 8 + 6];
      const float* n1b = bp[g * 8 + 7];
      k_prep_w2t<<<PREP_GRID, 256, 0, stream>>>(e1w, w2tg);
      k_lin128<<<BN * HID / 256, 256, 0, stream>>>(hcur, e0w, e0b, Pb);
      k_lin128<<<BN * HID / 256, 256, 0, stream>>>(hcur, e0w + HID * HID, nullptr, Qb);
      k_edge_msg<<<BN, 256, 0, stream>>>(Pb, Qb, dist, dist0,
                                         e0w + 256 * HID, e0w + 257 * HID,
                                         w2tg, e1b, em, aggb);
      k_node_a<<<BN * HID / 256, 256, 0, stream>>>(hcur, aggb, n0w, n0b, tmpb);
      k_node_b<<<BN * HID / 256, 256, 0, stream>>>(hcur, tmpb, n1w, n1b, nm, hnext);
      float* sw = hcur; hcur = hnext; hnext = sw;
    }
    const float* c0w = bp[16];
    const float* c0b = bp[17];
    const float* c1w = bp[18];
    const float* c1b = bp[19];
    const float* c2w = bp[20];
    k_prep_w2t<<<PREP_GRID, 256, 0, stream>>>(c1w, w2tg);
    k_lin128<<<BN * HID / 256, 256, 0, stream>>>(hcur, c0w, c0b, Pb);
    k_lin128<<<BN * HID / 256, 256, 0, stream>>>(hcur, c0w + HID * HID, nullptr, Qb);
    k_coord<<<BN, 256, 0, stream>>>(Pb, Qb, dist, dist0,
                                    c0w + 256 * HID, c0w + 257 * HID,
                                    w2tg, c1b, c2w, em, cd, xv);
  }
  k_out_h<<<BN * 8 / 256, 256, 0, stream>>>(hcur, out_w, out_b, nm, h8);
  k_finalize<<<BSZ, 64, 0, stream>>>(xv, x0v, h8, nm, (float*)d_out);
}
